// NeuralIntraAttentionModel_35218731827627
// MI455X (gfx1250) — matrix operand order repaired and checked by reading, not compiled, not run on hardware
//
#include <hip/hip_runtime.h>
#include <hip/hip_bf16.h>
#include <math.h>

#define BB 16
#define SS 512
#define TT 128
#define VV 32000
#define EE 128
#define HH 256
#define DD 512
#define VOC (VV - 3)          // 31997
#define KC  (3 * DD)          // 1536
#define MROWS (TT * BB)       // 2048, row r = t*BB + b
#define CHW 128               // vocab chunk width (8 waves * 16)
#define NCHUNK ((VOC + CHW - 1) / CHW)  // 250
#define CDIV(a, b) (((a) + (b) - 1) / (b))

typedef __attribute__((ext_vector_type(16))) __bf16 v16bf;
typedef __attribute__((ext_vector_type(8)))  float  v8f;
typedef unsigned int uint4v __attribute__((ext_vector_type(4)));

union V16 { v16bf v; uint4v q[2]; };

__device__ __forceinline__ float sigmoidf_(float x) { return 1.0f / (1.0f + expf(-x)); }

// ---- WMMA operand loaders (ISA 7.12.2 layouts, 16-bit, wave32) -------------
// A matrix 16x32: lane r(=lane&15) holds row r.
// lanes 0-15 : v[0..7]=K k0+0..7,  v[8..15]=K k0+16..23
// lanes16-31 : v[0..7]=K k0+8..15, v[8..15]=K k0+24..31
__device__ __forceinline__ v16bf load_tileA(const __hip_bfloat16* base, long ld,
                                            int rows_valid, int r, int hi, long k0) {
  V16 t;
  if (r < rows_valid) {
    const __hip_bfloat16* p0 = base + (long)r * ld + k0 + hi * 8;
    t.q[0] = *(const uint4v*)(p0);
    t.q[1] = *(const uint4v*)(p0 + 16);
  } else {
    t.q[0] = (uint4v)0u; t.q[1] = (uint4v)0u;
  }
  return t.v;
}
// B matrix 32x16 supplied as row-major N x K (i.e. B^T): lane n(=lane&15) holds col n.
// lanes 0-15 : v[0..7]=K k0+0..7, v[8..15]=K k0+16..23 ; lanes 16-31 : v[0..7]=K k0+8..15, v[8..15]=K k0+24..31
__device__ __forceinline__ v16bf load_tileB(const __hip_bfloat16* base, long ld,
                                            int rows_valid, int n, int hi, long k0) {
  V16 t;
  if (n < rows_valid) {
    const __hip_bfloat16* p0 = base + (long)n * ld + k0 + hi * 8;
    t.q[0] = *(const uint4v*)(p0);
    t.q[1] = *(const uint4v*)(p0 + 16);
  } else {
    t.q[0] = (uint4v)0u; t.q[1] = (uint4v)0u;
  }
  return t.v;
}

// ---- generic (optionally batched) GEMM: C[M,N] = A[M,K] * Bt[N,K]^T --------
// block = 256 thr = 8 waves, each wave owns one 16x16 C tile; block covers 128 cols.
template <int OUT_BF16, int ACT_TANH>
__global__ void gemm_tn(const __hip_bfloat16* __restrict__ A, long lda, long sA,
                        const __hip_bfloat16* __restrict__ Bt, long ldb, long sB,
                        void* __restrict__ Cv, long ldc, long sC,
                        const float* __restrict__ bias,
                        int M, int N, int K) {
  const int lane = threadIdx.x & 31;
  const int wave = threadIdx.x >> 5;
  const int r = lane & 15, hi = lane >> 4;
  const int n0 = blockIdx.x * 128 + wave * 16;
  const int m0 = blockIdx.y * 16;
  const long zb = blockIdx.z;
  if (m0 >= M) return;
  v8f acc = {};
  if (n0 < N) {
    const __hip_bfloat16* Ab = A + zb * sA + (long)m0 * lda;
    const __hip_bfloat16* Bb = Bt + zb * sB + (long)n0 * ldb;
    const int mrows = min(16, M - m0);
    const int nrows = min(16, N - n0);
    for (long k0 = 0; k0 < K; k0 += 32) {
      __builtin_prefetch(Bb + (long)r * ldb + k0 + 64, 0, 3);
      v16bf a = load_tileA(Ab, lda, mrows, r, hi, k0);
      v16bf b = load_tileB(Bb, ldb, nrows, r, hi, k0);
      acc = __builtin_amdgcn_wmma_f32_16x16x32_bf16(false, a, false, b, (short)0, acc,
                                                    false, false);
    }
    const int col = n0 + r;
    if (col < N) {
#pragma unroll
      for (int j = 0; j < 8; ++j) {
        int row = m0 + hi * 8 + j;
        if (row < M) {
          float v = acc[j];
          if (bias) v += bias[col];
          if (ACT_TANH) v = tanhf(v);
          long off = zb * sC + (long)row * ldc + col;
          if (OUT_BF16) ((__hip_bfloat16*)Cv)[off] = __float2bfloat16(v);
          else          ((float*)Cv)[off] = v;
        }
      }
    }
  }
}

// ---- LSTM recurrence (single workgroup, 512 thr = 16 waves) ----------------
// gates_t = gin[t] + h_{t-1} @ Whh^T + bias ; standard i,f,g,o cell.
__global__ void lstm_rec(const float* __restrict__ gin,            // [T,16,4*Hd]
                         const __hip_bfloat16* __restrict__ Whh,   // [4*Hd, Hd]
                         const float* __restrict__ bias,           // [4*Hd]
                         const float* __restrict__ h0,             // [16*Hd] f32 or null
                         __hip_bfloat16* __restrict__ hout,
                         long h_ss, long h_rs, long h_off,
                         float* __restrict__ hfin, long fin_rs,
                         int T, int Hd, int rev) {
  extern __shared__ char smem[];
  const int N4 = 4 * Hd;
  float* gsh = (float*)smem;                                   // 16 x N4
  __hip_bfloat16* hsh = (__hip_bfloat16*)(smem + (size_t)16 * N4 * sizeof(float));
  const int tid = threadIdx.x;
  const int lane = tid & 31, wave = tid >> 5;
  const int r = lane & 15, hi = lane >> 4;
  const int tpw = (N4 / 16) / 16;     // N tiles per wave (4 enc / 8 dec)
  const int ncell = (16 * Hd) / 512;  // cells per thread (8 enc / 16 dec)
  float creg[16];
#pragma unroll
  for (int i = 0; i < 16; ++i) creg[i] = 0.0f;
  for (int idx = tid; idx < 16 * Hd; idx += 512)
    hsh[idx] = __float2bfloat16(h0 ? h0[idx] : 0.0f);
  __syncthreads();
  for (int s = 0; s < T; ++s) {
    const int t = rev ? (T - 1 - s) : s;
    for (int i = 0; i < tpw; ++i) {
      const int n0 = (wave * tpw + i) * 16;
      v8f acc = {};
      for (int k0 = 0; k0 < Hd; k0 += 32) {
        v16bf a = load_tileA(hsh, Hd, 16, r, hi, k0);
        v16bf b = load_tileB(Whh + (long)n0 * Hd, Hd, 16, r, hi, k0);
        acc = __builtin_amdgcn_wmma_f32_16x16x32_bf16(false, a, false, b, (short)0, acc,
                                                      false, false);
      }
      const int col = n0 + r;
#pragma unroll
      for (int j = 0; j < 8; ++j) gsh[(hi * 8 + j) * N4 + col] = acc[j];
    }
    __syncthreads();
    for (int i = 0; i < ncell; ++i) {
      const int cell = tid + i * 512;
      const int b = cell / Hd, j = cell % Hd;
      const float* gr = gin + ((long)t * 16 + b) * N4;
      const float* gs = gsh + (long)b * N4;
      float gi = gs[j]          + gr[j]          + bias[j];
      float gf = gs[Hd + j]     + gr[Hd + j]     + bias[Hd + j];
      float gg = gs[2 * Hd + j] + gr[2 * Hd + j] + bias[2 * Hd + j];
      float go = gs[3 * Hd + j] + gr[3 * Hd + j] + bias[3 * Hd + j];
      float c = sigmoidf_(gf) * creg[i] + sigmoidf_(gi) * tanhf(gg);
      float h = sigmoidf_(go) * tanhf(c);
      creg[i] = c;
      __hip_bfloat16 hb = __float2bfloat16(h);
      hsh[(long)b * Hd + j] = hb;
      hout[h_off + (long)t * h_ss + (long)b * h_rs + j] = hb;
      if (hfin && s == T - 1) hfin[(long)b * fin_rs + j] = h;
    }
    __syncthreads();
  }
}

// ---- fused vocab GEMM + chunked softmax stats ------------------------------
// block: 8 waves -> 16 rows x 128 vocab cols. The 16-row A block (48 KB bf16)
// is staged into LDS once per block via CDNA5 async global->LDS copies
// (ASYNCcnt path), then WMMA A operands come from LDS while B (out_proj,
// L2-resident) streams from global. Epilogue: per-chunk max / sum-exp and the
// logit at each row's teacher-token index (online softmax merge done later).
__global__ void vocab_fused(const __hip_bfloat16* __restrict__ A,   // [MROWS,KC] concat
                            const __hip_bfloat16* __restrict__ Bt,  // [VOC,KC] out_proj
                            const float* __restrict__ ob,           // [VOC]
                            const int* __restrict__ tgt,            // [BB,TT]
                            float* __restrict__ chmax, float* __restrict__ chsum,
                            float* __restrict__ tlog) {
  __shared__ __align__(16) __hip_bfloat16 ash[16 * KC];  // 48 KB staged A block
  __shared__ float tile[16 * CHW];                       // 8 KB
  const int tid = threadIdx.x;
  const int lane = tid & 31, wave = tid >> 5;
  const int r = lane & 15, hi = lane >> 4;
  const int nbase = blockIdx.x * CHW;
  const int n0 = nbase + wave * 16;
  const int m0 = blockIdx.y * 16;
  const int nrows = max(0, min(16, VOC - n0));

  // -- async stage: 16 contiguous A rows (16*KC*2 = 49152 B) -> LDS ----------
  {
    const char* gsrc = (const char*)(A + (long)m0 * KC);
    const unsigned ldsbase = (unsigned)(uintptr_t)(void*)ash;  // low 32b = LDS offset
    const int nchunks = (16 * KC * 2) / 16;                    // 3072 x 16B
    for (int i = tid; i < nchunks; i += 256) {
      unsigned ldst = ldsbase + (unsigned)i * 16u;
      const char* g = gsrc + (long)i * 16;
      asm volatile("global_load_async_to_lds_b128 %0, %1, off"
                   :: "v"(ldst), "v"(g) : "memory");
    }
    asm volatile("s_wait_asynccnt 0x0" ::: "memory");
  }
  __syncthreads();

  v8f acc = {};
  if (nrows > 0) {
    const __hip_bfloat16* Bb = Bt + (long)n0 * KC;
    for (long k0 = 0; k0 < KC; k0 += 32) {
      __builtin_prefetch(Bb + (long)r * KC + k0 + 64, 0, 3);
      v16bf a = load_tileA(ash, KC, 16, r, hi, k0);
      v16bf b = load_tileB(Bb, KC, nrows, r, hi, k0);
      acc = __builtin_amdgcn_wmma_f32_16x16x32_bf16(false, a, false, b, (short)0, acc,
                                                    false, false);
    }
  }
  const int gcol = n0 + r;
#pragma unroll
  for (int j = 0; j < 8; ++j) {
    float v = (nrows > 0 && gcol < VOC) ? (acc[j] + ob[gcol]) : -INFINITY;
    tile[(hi * 8 + j) * CHW + wave * 16 + r] = v;
  }
  __syncthreads();
  if (tid < 16) {
    const int row = m0 + tid;
    float mx = -INFINITY;
    for (int c = 0; c < CHW; ++c) mx = fmaxf(mx, tile[tid * CHW + c]);
    float sm = 0.0f;
    for (int c = 0; c < CHW; ++c) sm += expf(tile[tid * CHW + c] - mx);
    chmax[(long)row * NCHUNK + blockIdx.x] = mx;
    chsum[(long)row * NCHUNK + blockIdx.x] = sm;
    const int b = row & (BB - 1);
    const int t = row / BB;
    int nt = tgt[b * TT + t];
    int idx = min(max(nt - 3, 0), VOC - 1);
    if (idx >= nbase && idx < nbase + CHW) tlog[row] = tile[tid * CHW + (idx - nbase)];
  }
}

// ---- small elementwise / reduction kernels ---------------------------------
__global__ void f32_to_bf16_k(const float* __restrict__ in, __hip_bfloat16* __restrict__ out, long n) {
  long i = (long)blockIdx.x * blockDim.x + threadIdx.x;
  if (i < n) out[i] = __float2bfloat16(in[i]);
}
__global__ void trans_f32_bf16_k(const float* __restrict__ in, __hip_bfloat16* __restrict__ out,
                                 int R, int C) {  // out[C,R] = in[R,C]^T
  long i = (long)blockIdx.x * blockDim.x + threadIdx.x;
  if (i < (long)R * C) {
    int r = (int)(i / C), c = (int)(i % C);
    out[(long)c * R + r] = __float2bfloat16(in[i]);
  }
}
__global__ void trans_bsd_bds_k(const __hip_bfloat16* __restrict__ in, __hip_bfloat16* __restrict__ out) {
  long i = (long)blockIdx.x * blockDim.x + threadIdx.x;   // [BB,SS,DD] -> [BB,DD,SS]
  if (i < (long)BB * SS * DD) {
    int b = (int)(i / ((long)SS * DD));
    long rem = i % ((long)SS * DD);
    int s = (int)(rem / DD), d = (int)(rem % DD);
    out[(long)b * DD * SS + (long)d * SS + s] = in[i];
  }
}
__global__ void trans_tbd_bdt_k(const __hip_bfloat16* __restrict__ in, __hip_bfloat16* __restrict__ out) {
  long i = (long)blockIdx.x * blockDim.x + threadIdx.x;   // [TT,BB,DD] -> [BB,DD,TT]
  if (i < (long)TT * BB * DD) {
    int t = (int)(i / ((long)BB * DD));
    long rem = i % ((long)BB * DD);
    int b = (int)(rem / DD), d = (int)(rem % DD);
    out[(long)b * DD * TT + (long)d * TT + t] = in[i];
  }
}
__global__ void gather_enc_k(const __hip_bfloat16* __restrict__ embb, const int* __restrict__ ids,
                             __hip_bfloat16* __restrict__ x) {  // x [SS,BB,EE]
  long i = (long)blockIdx.x * blockDim.x + threadIdx.x;
  if (i < (long)SS * BB * EE) {
    int s = (int)(i / (BB * EE));
    long rem = i % (BB * EE);
    int b = (int)(rem / EE), e = (int)(rem % EE);
    int id = ids[b * SS + s];
    if (id >= VV || id < 0) id = 1;  // UNK
    x[i] = embb[(long)id * EE + e];
  }
}
__global__ void gather_dec_k(const __hip_bfloat16* __restrict__ embb, const int* __restrict__ tgt,
                             __hip_bfloat16* __restrict__ x) {  // x [TT,BB,EE]
  long i = (long)blockIdx.x * blockDim.x + threadIdx.x;
  if (i < (long)TT * BB * EE) {
    int t = (int)(i / (BB * EE));
    long rem = i % (BB * EE);
    int b = (int)(rem / EE), e = (int)(rem % EE);
    int id = (t == 0) ? 2 : tgt[b * TT + (t - 1)];  // START = 2
    if (id >= VV || id < 0) id = 1;
    x[i] = embb[(long)id * EE + e];
  }
}
// scores -> temporal (e / cum_prev), sequential prefix over t per (b,s) column
__global__ void exp_cum_kernel(float* __restrict__ sc, const int* __restrict__ ids) {
  const int idx = blockIdx.x * blockDim.x + threadIdx.x;  // BB*SS
  if (idx >= BB * SS) return;
  const int b = idx / SS, s = idx % SS;
  const bool pad = (ids[b * SS + s] == 0);
  float cum = 0.0f;
  for (int t = 0; t < TT; ++t) {
    const long o = ((long)b * TT + t) * SS + s;
    float e = expf(fminf(sc[o], 30.0f));
    float tm = e / (t == 0 ? 1.0f : (cum + 1e-8f));
    sc[o] = pad ? -INFINITY : tm;
    cum += e;
  }
}
// softmax over S of temporal; also emits copy_p[row] = sum attn * (input==nt)
__global__ void attn_softmax(const float* __restrict__ tmprl, const int* __restrict__ ids,
                             const int* __restrict__ tgt, __hip_bfloat16* __restrict__ attn_bf,
                             float* __restrict__ copyp) {
  __shared__ float red[256];
  const int blk = blockIdx.x;  // b*TT + t
  const int b = blk / TT, t = blk % TT;
  const long base = (long)blk * SS;
  const int tid = threadIdx.x;
  float mx = -INFINITY;
  for (int s = tid; s < SS; s += 256) mx = fmaxf(mx, tmprl[base + s]);
  red[tid] = mx; __syncthreads();
  for (int o = 128; o > 0; o >>= 1) { if (tid < o) red[tid] = fmaxf(red[tid], red[tid + o]); __syncthreads(); }
  mx = red[0]; __syncthreads();
  float sm = 0.0f;
  for (int s = tid; s < SS; s += 256) sm += expf(tmprl[base + s] - mx);
  red[tid] = sm; __syncthreads();
  for (int o = 128; o > 0; o >>= 1) { if (tid < o) red[tid] += red[tid + o]; __syncthreads(); }
  sm = red[0]; __syncthreads();
  const float inv = 1.0f / sm;
  const int nt = tgt[b * TT + t];
  float cp = 0.0f;
  for (int s = tid; s < SS; s += 256) {
    float a = expf(tmprl[base + s] - mx) * inv;
    attn_bf[base + s] = __float2bfloat16(a);
    if (ids[b * SS + s] == nt) cp += a;
  }
  red[tid] = cp; __syncthreads();
  for (int o = 128; o > 0; o >>= 1) { if (tid < o) red[tid] += red[tid + o]; __syncthreads(); }
  if (tid == 0) copyp[t * BB + b] = red[0];
}
// intra-decoder softmax over u (<t masked with -1e9); t==0 -> all-zero row
__global__ void dec_softmax(const float* __restrict__ dsc, __hip_bfloat16* __restrict__ smo) {
  __shared__ float red[TT];
  const int blk = blockIdx.x;  // b*TT + t
  const int t = blk % TT;
  const long base = (long)blk * TT;
  const int u = threadIdx.x;
  if (t == 0) { smo[base + u] = __float2bfloat16(0.0f); return; }
  float x = (u < t) ? dsc[base + u] : -1e9f;
  red[u] = x; __syncthreads();
  for (int o = TT / 2; o > 0; o >>= 1) { if (u < o) red[u] = fmaxf(red[u], red[u + o]); __syncthreads(); }
  float mx = red[0]; __syncthreads();
  float e = expf(x - mx);
  red[u] = e; __syncthreads();
  for (int o = TT / 2; o > 0; o >>= 1) { if (u < o) red[u] += red[u + o]; __syncthreads(); }
  smo[base + u] = __float2bfloat16(e / red[0]);
}
__global__ void build_concat_k(const __hip_bfloat16* __restrict__ h1, const float* __restrict__ ectx,
                               const float* __restrict__ dctx, float* __restrict__ cf,
                               __hip_bfloat16* __restrict__ cb) {
  long i = (long)blockIdx.x * blockDim.x + threadIdx.x;
  if (i < (long)MROWS * KC) {
    int r = (int)(i / KC), c = (int)(i % KC);
    int t = r / BB, b = r % BB;
    float v;
    if (c < DD)            v = __bfloat162float(h1[(long)t * BB * DD + (long)b * DD + c]);
    else if (c < 2 * DD)   v = ectx[((long)b * TT + t) * DD + (c - DD)];
    else                   v = dctx[((long)b * TT + t) * DD + (c - 2 * DD)];
    cf[i] = v;
    cb[i] = __float2bfloat16(v);
  }
}
__global__ void pgen_kernel(const float* __restrict__ cf, const float* __restrict__ sW,
                            const float* __restrict__ sb, float* __restrict__ pg) {
  __shared__ float red[256];
  const int row = blockIdx.x;
  const int tid = threadIdx.x;
  float s = 0.0f;
  for (int k = tid; k < KC; k += 256) s += cf[(long)row * KC + k] * sW[k];
  red[tid] = s; __syncthreads();
  for (int o = 128; o > 0; o >>= 1) { if (tid < o) red[tid] += red[tid + o]; __syncthreads(); }
  if (tid == 0) pg[row] = sigmoidf_(red[0] + sb[0]);
}
__global__ void nll_kernel(const float* __restrict__ chmax, const float* __restrict__ chsum,
                           const float* __restrict__ tlog, const float* __restrict__ pg,
                           const float* __restrict__ cp, const int* __restrict__ tgt,
                           const int* __restrict__ tlen, float* __restrict__ out) {
  __shared__ float red[TT];
  const int b = blockIdx.x;
  const int t = threadIdx.x;
  const int row = t * BB + b;
  float mx = -INFINITY;
  for (int c = 0; c < NCHUNK; ++c) mx = fmaxf(mx, chmax[(long)row * NCHUNK + c]);
  float Z = 0.0f;
  for (int c = 0; c < NCHUNK; ++c)
    Z += chsum[(long)row * NCHUNK + c] * expf(chmax[(long)row * NCHUNK + c] - mx);
  const int nt = tgt[b * TT + t];
  float gen = (nt >= 3 && nt < VV) ? expf(tlog[row] - mx) / Z : 0.0f;
  float p = pg[row] * gen + (1.0f - pg[row]) * cp[row];
  float term = (t < tlen[b]) ? -logf(p + 1e-9f) : 0.0f;
  red[t] = term; __syncthreads();
  for (int o = TT / 2; o > 0; o >>= 1) { if (t < o) red[t] += red[t + o]; __syncthreads(); }
  if (t == 0) out[b] = red[0];
}

// ---------------------------------------------------------------------------
extern "C" void kernel_launch(void* const* d_in, const int* in_sizes, int n_in,
                              void* d_out, int out_size, void* d_ws, size_t ws_size,
                              hipStream_t stream) {
  (void)in_sizes; (void)n_in; (void)out_size; (void)ws_size;
  const int* input_ids  = (const int*)d_in[0];
  const int* target_ids = (const int*)d_in[1];
  const int* tlen       = (const int*)d_in[3];
  const float* emb    = (const float*)d_in[5];
  const float* eWih0f = (const float*)d_in[6],  *eWhh0f = (const float*)d_in[7],  *eb0f = (const float*)d_in[8];
  const float* eWih0b = (const float*)d_in[9],  *eWhh0b = (const float*)d_in[10], *eb0b = (const float*)d_in[11];
  const float* eWih1f = (const float*)d_in[12], *eWhh1f = (const float*)d_in[13], *eb1f = (const float*)d_in[14];
  const float* eWih1b = (const float*)d_in[15], *eWhh1b = (const float*)d_in[16], *eb1b = (const float*)d_in[17];
  const float* dWih0 = (const float*)d_in[18], *dWhh0 = (const float*)d_in[19], *db0 = (const float*)d_in[20];
  const float* dWih1 = (const float*)d_in[21], *dWhh1 = (const float*)d_in[22], *db1 = (const float*)d_in[23];
  const float* Wenc = (const float*)d_in[24];
  const float* Wdec = (const float*)d_in[25];
  const float* Wvoc = (const float*)d_in[26];
  const float* sW   = (const float*)d_in[27];
  const float* sb   = (const float*)d_in[28];
  const float* ob   = (const float*)d_in[29];

  char* wp = (char*)d_ws;
  auto alloc = [&](size_t bytes) -> char* {
    char* p = wp; wp += (bytes + 255) & ~(size_t)255; return p;
  };
  auto conv = [&](const float* src, long n) -> __hip_bfloat16* {
    __hip_bfloat16* dst = (__hip_bfloat16*)alloc((size_t)n * 2);
    f32_to_bf16_k<<<dim3(CDIV(n, 256)), dim3(256), 0, stream>>>(src, dst, n);
    return dst;
  };
  auto gemm_f32 = [&](const __hip_bfloat16* A, long lda, long sA,
                      const __hip_bfloat16* Bt, long ldb, long sB,
                      float* C, long ldc, long sC, int M, int N, int K, int nb) {
    gemm_tn<0, 0><<<dim3(CDIV(N, 128), CDIV(M, 16), nb), dim3(256), 0, stream>>>(
        A, lda, sA, Bt, ldb, sB, (void*)C, ldc, sC, nullptr, M, N, K);
  };
  auto gemm_bf = [&](const __hip_bfloat16* A, long lda, const __hip_bfloat16* Bt, long ldb,
                     __hip_bfloat16* C, long ldc, int M, int N, int K) {
    gemm_tn<1, 0><<<dim3(CDIV(N, 128), CDIV(M, 16), 1), dim3(256), 0, stream>>>(
        A, lda, 0, Bt, ldb, 0, (void*)C, ldc, 0, nullptr, M, N, K);
  };

  // ---- weight prep (bf16, transposes where GEMM needs N x K) ----
  __hip_bfloat16* emb_bf = conv(emb, (long)VV * EE);
  __hip_bfloat16* bWih0f = conv(eWih0f, 4 * HH * EE);
  __hip_bfloat16* bWhh0f = conv(eWhh0f, 4 * HH * HH);
  __hip_bfloat16* bWih0b = conv(eWih0b, 4 * HH * EE);
  __hip_bfloat16* bWhh0b = conv(eWhh0b, 4 * HH * HH);
  __hip_bfloat16* bWih1f = conv(eWih1f, 4 * HH * DD);
  __hip_bfloat16* bWhh1f = conv(eWhh1f, 4 * HH * HH);
  __hip_bfloat16* bWih1b = conv(eWih1b, 4 * HH * DD);
  __hip_bfloat16* bWhh1b = conv(eWhh1b, 4 * HH * HH);
  __hip_bfloat16* bdWih0 = conv(dWih0, 4 * DD * EE);
  __hip_bfloat16* bdWhh0 = conv(dWhh0, 4 * DD * DD);
  __hip_bfloat16* bdWih1 = conv(dWih1, 4 * DD * DD);
  __hip_bfloat16* bdWhh1 = conv(dWhh1, 4 * DD * DD);
  __hip_bfloat16* WencT = (__hip_bfloat16*)alloc((size_t)DD * DD * 2);
  trans_f32_bf16_k<<<dim3(CDIV(DD * DD, 256)), dim3(256), 0, stream>>>(Wenc, WencT, DD, DD);
  __hip_bfloat16* WdecT = (__hip_bfloat16*)alloc((size_t)DD * DD * 2);
  trans_f32_bf16_k<<<dim3(CDIV(DD * DD, 256)), dim3(256), 0, stream>>>(Wdec, WdecT, DD, DD);
  __hip_bfloat16* WvocT = (__hip_bfloat16*)alloc((size_t)KC * EE * 2);
  trans_f32_bf16_k<<<dim3(CDIV(EE * KC, 256)), dim3(256), 0, stream>>>(Wvoc, WvocT, EE, KC);

  // ---- encoder ----
  __hip_bfloat16* x_enc = (__hip_bfloat16*)alloc((size_t)SS * BB * EE * 2);
  gather_enc_k<<<dim3(CDIV((long)SS * BB * EE, 256)), dim3(256), 0, stream>>>(emb_bf, input_ids, x_enc);
  float* g_f = (float*)alloc((size_t)SS * BB * 4 * HH * 4);  // reused for dec gates later
  float* g_b = (float*)alloc((size_t)SS * BB * 4 * HH * 4);
  gemm_f32(x_enc, EE, 0, bWih0f, EE, 0, g_f, 4 * HH, 0, SS * BB, 4 * HH, EE, 1);
  gemm_f32(x_enc, EE, 0, bWih0b, EE, 0, g_b, 4 * HH, 0, SS * BB, 4 * HH, EE, 1);

  __hip_bfloat16* x1 = (__hip_bfloat16*)alloc((size_t)SS * BB * DD * 2);  // [S,B,2H]
  float* decinit0 = (float*)alloc((size_t)BB * DD * 4);
  float* decinit1 = (float*)alloc((size_t)BB * DD * 4);
  const size_t sh_enc = (size_t)16 * 4 * HH * 4 + (size_t)16 * HH * 2;
  const size_t sh_dec = (size_t)16 * 4 * DD * 4 + (size_t)16 * DD * 2;
  lstm_rec<<<1, 512, sh_enc, stream>>>(g_f, bWhh0f, eb0f, nullptr, x1, (long)BB * DD, DD, 0,
                                       decinit0, DD, SS, HH, 0);
  lstm_rec<<<1, 512, sh_enc, stream>>>(g_b, bWhh0b, eb0b, nullptr, x1, (long)BB * DD, DD, HH,
                                       decinit0 + HH, DD, SS, HH, 1);
  gemm_f32(x1, DD, 0, bWih1f, DD, 0, g_f, 4 * HH, 0, SS * BB, 4 * HH, DD, 1);
  gemm_f32(x1, DD, 0, bWih1b, DD, 0, g_b, 4 * HH, 0, SS * BB, 4 * HH, DD, 1);
  __hip_bfloat16* enc_hs = (__hip_bfloat16*)alloc((size_t)BB * SS * DD * 2);  // [B,S,D]
  lstm_rec<<<1, 512, sh_enc, stream>>>(g_f, bWhh1f, eb1f, nullptr, enc_hs, DD, (long)SS * DD, 0,
                                       decinit1, DD, SS, HH, 0);
  lstm_rec<<<1, 512, sh_enc, stream>>>(g_b, bWhh1b, eb1b, nullptr, enc_hs, DD, (long)SS * DD, HH,
                                       decinit1 + HH, DD, SS, HH, 1);
  __hip_bfloat16* enc_hsT = (__hip_bfloat16*)alloc((size_t)BB * DD * SS * 2);  // [B,D,S]
  trans_bsd_bds_k<<<dim3(CDIV((long)BB * SS * DD, 256)), dim3(256), 0, stream>>>(enc_hs, enc_hsT);

  // ---- decoder LSTM (inputs known under teacher forcing) ----
  __hip_bfloat16* x_dec = (__hip_bfloat16*)alloc((size_t)TT * BB * EE * 2);
  gather_dec_k<<<dim3(CDIV((long)TT * BB * EE, 256)), dim3(256), 0, stream>>>(emb_bf, target_ids, x_dec);
  gemm_f32(x_dec, EE, 0, bdWih0, EE, 0, g_f, 4 * DD, 0, MROWS, 4 * DD, EE, 1);
  __hip_bfloat16* h0d = (__hip_bfloat16*)alloc((size_t)TT * BB * DD * 2);
  lstm_rec<<<1, 512, sh_dec, stream>>>(g_f, bdWhh0, db0, decinit0, h0d, (long)BB * DD, DD, 0,
                                       nullptr, 0, TT, DD, 0);
  gemm_f32(h0d, DD, 0, bdWih1, DD, 0, g_b, 4 * DD, 0, MROWS, 4 * DD, DD, 1);
  __hip_bfloat16* h1a = (__hip_bfloat16*)alloc((size_t)TT * BB * DD * 2);  // [T,B,D]
  lstm_rec<<<1, 512, sh_dec, stream>>>(g_b, bdWhh1, db1, decinit1, h1a, (long)BB * DD, DD, 0,
                                       nullptr, 0, TT, DD, 0);
  __hip_bfloat16* h1T = (__hip_bfloat16*)alloc((size_t)BB * DD * TT * 2);  // [B,D,T]
  trans_tbd_bdt_k<<<dim3(CDIV((long)TT * BB * DD, 256)), dim3(256), 0, stream>>>(h1a, h1T);

  // ---- intra-temporal encoder attention (batched over all T) ----
  __hip_bfloat16* Qe = (__hip_bfloat16*)alloc((size_t)MROWS * DD * 2);
  gemm_bf(h1a, DD, WencT, DD, Qe, DD, MROWS, DD, DD);
  float* scores = (float*)alloc((size_t)BB * TT * SS * 4);
  gemm_f32(Qe, (long)BB * DD, DD, enc_hs, DD, (long)SS * DD, scores, SS, (long)TT * SS,
           TT, SS, DD, BB);
  exp_cum_kernel<<<dim3(CDIV(BB * SS, 256)), dim3(256), 0, stream>>>(scores, input_ids);
  __hip_bfloat16* attn_bf = (__hip_bfloat16*)alloc((size_t)BB * TT * SS * 2);
  float* copyp = (float*)alloc((size_t)MROWS * 4);
  attn_softmax<<<dim3(BB * TT), dim3(256), 0, stream>>>(scores, input_ids, target_ids, attn_bf, copyp);
  float* ectx = (float*)alloc((size_t)BB * TT * DD * 4);
  gemm_f32(attn_bf, SS, (long)TT * SS, enc_hsT, SS, (long)DD * SS, ectx, DD, (long)TT * DD,
           TT, DD, SS, BB);

  // ---- intra-decoder attention ----
  __hip_bfloat16* Qd = (__hip_bfloat16*)alloc((size_t)MROWS * DD * 2);
  gemm_bf(h1a, DD, WdecT, DD, Qd, DD, MROWS, DD, DD);
  float* dsc = (float*)alloc((size_t)BB * TT * TT * 4);
  gemm_f32(Qd, (long)BB * DD, DD, h1a, (long)BB * DD, DD, dsc, TT, (long)TT * TT,
           TT, TT, DD, BB);
  __hip_bfloat16* sm_bf = (__hip_bfloat16*)alloc((size_t)BB * TT * TT * 2);
  dec_softmax<<<dim3(BB * TT), dim3(TT), 0, stream>>>(dsc, sm_bf);
  float* dctx = (float*)alloc((size_t)BB * TT * DD * 4);
  gemm_f32(sm_bf, TT, (long)TT * TT, h1T, TT, (long)DD * TT, dctx, DD, (long)TT * DD,
           TT, DD, TT, BB);

  // ---- output head ----
  float* concat_f = (float*)alloc((size_t)MROWS * KC * 4);
  __hip_bfloat16* concat_b = (__hip_bfloat16*)alloc((size_t)MROWS * KC * 2);
  build_concat_k<<<dim3(CDIV((long)MROWS * KC, 256)), dim3(256), 0, stream>>>(h1a, ectx, dctx,
                                                                              concat_f, concat_b);
  float* pg = (float*)alloc((size_t)MROWS * 4);
  pgen_kernel<<<dim3(MROWS), dim3(256), 0, stream>>>(concat_f, sW, sb, pg);

  __hip_bfloat16* outproj = (__hip_bfloat16*)alloc((size_t)VOC * KC * 2);  // tanh(emb[3:] @ Wvoc)
  gemm_tn<1, 1><<<dim3(CDIV(KC, 128), CDIV(VOC, 16), 1), dim3(256), 0, stream>>>(
      emb_bf + 3 * EE, EE, 0, WvocT, EE, 0, (void*)outproj, KC, 0, nullptr, VOC, KC, EE);

  float* chmax = (float*)alloc((size_t)MROWS * NCHUNK * 4);
  float* chsum = (float*)alloc((size_t)MROWS * NCHUNK * 4);
  float* tlog  = (float*)alloc((size_t)MROWS * 4);
  vocab_fused<<<dim3(NCHUNK, MROWS / 16), dim3(256), 0, stream>>>(concat_b, outproj, ob,
                                                                  target_ids, chmax, chsum, tlog);
  nll_kernel<<<dim3(BB), dim3(TT), 0, stream>>>(chmax, chsum, tlog, pg, copyp, target_ids, tlen,
                                                (float*)d_out);
}
